// get_model_49529562857492
// MI455X (gfx1250) — compile-verified, matrix operand order checked by reading, not run on hardware
//
#include <hip/hip_runtime.h>

// ---------------------------------------------------------------------------
// Types for CDNA5 WMMA (wave32): v_wmma_f32_16x16x32_bf16
// ---------------------------------------------------------------------------
typedef __bf16 bf16_t;
typedef bf16_t v16bf __attribute__((ext_vector_type(16)));
typedef float  v8f   __attribute__((ext_vector_type(8)));

union Frag { v16bf v; unsigned u[8]; };

__device__ __forceinline__ unsigned short f2bf_u(float f) {
  unsigned u = __float_as_uint(f);
  unsigned r = u + 0x7FFFu + ((u >> 16) & 1u);   // round-to-nearest-even
  return (unsigned short)(r >> 16);
}
__device__ __forceinline__ float bfu2f(unsigned short s) {
  return __uint_as_float(((unsigned)s) << 16);
}
__device__ __forceinline__ int imin(int a, int b) { return a < b ? a : b; }

#define CEILDIV(a, b) (((a) + (b) - 1) / (b))
#define RUP32(x) (((x) + 31) & ~31)
#define GEMM_WAVES 4
#define NSUB 4   // 16-row subtiles per wave -> 64 rows x 16 cols per wave

// ---------------------------------------------------------------------------
// Weight pre-pack: fp32 W[Cin,Cout] -> bf16 Wp[ldc][ldk], transposed (n-major,
// k-contiguous) and zero-padded, so the GEMM B staging is guard-free uint2s.
// ---------------------------------------------------------------------------
__global__ void pack_w_kernel(const float* __restrict__ W, unsigned short* __restrict__ Wp,
                              int Cin, int Cout, int ldk, int total) {
  int t = blockIdx.x * blockDim.x + threadIdx.x;
  if (t >= total) return;
  int k = t % ldk, n = t / ldk;
  float v = (n < Cout && k < Cin) ? W[(size_t)k * Cout + n] : 0.f;
  Wp[(size_t)n * ldk + k] = f2bf_u(v);
}

// ---------------------------------------------------------------------------
// Input prep: [B,6,N] fp32 -> l0_xyz [B,N,3] fp32 + l0_pts [B,N,6] bf16
// ---------------------------------------------------------------------------
__global__ void prep_kernel(const float* __restrict__ xin, float* __restrict__ oxyz,
                            unsigned short* __restrict__ opts, int N, int total) {
  int t = blockIdx.x * blockDim.x + threadIdx.x;
  if (t >= total) return;
  int n = t % N, b = t / N;
  const float* src = xin + (size_t)b * 6 * N;
  float* ox = oxyz + (size_t)t * 3;
  unsigned short* op = opts + (size_t)t * 6;
  for (int c = 0; c < 3; ++c) ox[c] = src[(size_t)c * N + n];
  for (int c = 0; c < 6; ++c) op[c] = f2bf_u(src[(size_t)c * N + n]);
}

// ---------------------------------------------------------------------------
// Farthest point sampling: one workgroup per batch, dist array in LDS
// ---------------------------------------------------------------------------
__global__ void fps_kernel(const float* __restrict__ xyz, int N, int npoint,
                           int* __restrict__ out) {
  extern __shared__ float smem[];
  float* dist = smem;                       // N floats
  float* rval = smem + N;                   // blockDim floats
  int*   ridx = (int*)(rval + blockDim.x);  // blockDim ints
  int b = blockIdx.x;
  const float* P = xyz + (size_t)b * N * 3;
  for (int n = threadIdx.x; n < N; n += blockDim.x) dist[n] = 1e10f;
  __syncthreads();
  int far = 0;
  for (int i = 0; i < npoint; ++i) {
    if (threadIdx.x == 0) out[b * npoint + i] = far;
    float cx = P[far * 3], cy = P[far * 3 + 1], cz = P[far * 3 + 2];
    float bm = -1.f; int bi = 0;
    for (int n = threadIdx.x; n < N; n += blockDim.x) {
      float dx = P[n * 3] - cx, dy = P[n * 3 + 1] - cy, dz = P[n * 3 + 2] - cz;
      float dn = fminf(dist[n], dx * dx + dy * dy + dz * dz);
      dist[n] = dn;
      if (dn > bm) { bm = dn; bi = n; }
    }
    rval[threadIdx.x] = bm; ridx[threadIdx.x] = bi;
    __syncthreads();
    for (int s = blockDim.x / 2; s > 0; s >>= 1) {
      if (threadIdx.x < (unsigned)s) {
        float ov = rval[threadIdx.x + s]; int oi = ridx[threadIdx.x + s];
        float mv = rval[threadIdx.x];     int mi = ridx[threadIdx.x];
        if (ov > mv || (ov == mv && oi < mi)) { rval[threadIdx.x] = ov; ridx[threadIdx.x] = oi; }
      }
      __syncthreads();
    }
    far = ridx[0];
    __syncthreads();
  }
}

__global__ void gather_xyz_kernel(const float* __restrict__ xyz, const int* __restrict__ idx,
                                  float* __restrict__ out, int Nin, int np, int total) {
  int t = blockIdx.x * blockDim.x + threadIdx.x;
  if (t >= total) return;
  int m = t % np, b = t / np;
  int src = idx[b * np + m];
  const float* p = xyz + ((size_t)b * Nin + src) * 3;
  float* o = out + (size_t)t * 3;
  o[0] = p[0]; o[1] = p[1]; o[2] = p[2];
}

// ---------------------------------------------------------------------------
// Ball query: first K indices (in original order) within radius, pad w/ first
// ---------------------------------------------------------------------------
__global__ void ball_query_kernel(const float* __restrict__ xyz, const float* __restrict__ nxyz,
                                  int* __restrict__ gidx, int N, int S, int K, float r2,
                                  int total) {
  int t = blockIdx.x * blockDim.x + threadIdx.x;
  if (t >= total) return;
  int m = t % S, b = t / S;
  const float* c = nxyz + (size_t)t * 3;
  const float* P = xyz + (size_t)b * N * 3;
  int* o = gidx + ((size_t)b * S + m) * K;
  int cnt = 0;
  for (int n = 0; n < N && cnt < K; ++n) {
    float dx = P[n * 3] - c[0], dy = P[n * 3 + 1] - c[1], dz = P[n * 3 + 2] - c[2];
    if (dx * dx + dy * dy + dz * dz <= r2) o[cnt++] = n;
  }
  int pad = (cnt > 0) ? o[0] : 0;
  for (int k = cnt; k < K; ++k) o[k] = pad;
}

// ---------------------------------------------------------------------------
// Grouping: X[b*S*K, lda] = [pts[gidx] (bf16), xyz[gidx]-center, zero pad]
// lda = RUP32(C+3): rows zero-padded so GEMM needs no K guards
// ---------------------------------------------------------------------------
__global__ void group_kernel(const unsigned short* __restrict__ pts, const float* __restrict__ xyz,
                             const float* __restrict__ nxyz, const int* __restrict__ gidx,
                             unsigned short* __restrict__ X, int Npts, int S, int K, int C,
                             int lda, int total) {
  int t = blockIdx.x * blockDim.x + threadIdx.x;
  if (t >= total) return;
  int k = t % K, m = (t / K) % S, b = t / (K * S);
  int src = gidx[((size_t)b * S + m) * K + k];
  const unsigned short* pr = pts + ((size_t)b * Npts + src) * C;
  unsigned short* orow = X + (size_t)t * lda;
  for (int c = 0; c < C; ++c) orow[c] = pr[c];
  const float* p = xyz + ((size_t)b * Npts + src) * 3;
  const float* cen = nxyz + ((size_t)b * S + m) * 3;
  orow[C]     = f2bf_u(p[0] - cen[0]);
  orow[C + 1] = f2bf_u(p[1] - cen[1]);
  orow[C + 2] = f2bf_u(p[2] - cen[2]);
  for (int c = C + 3; c < lda; ++c) orow[c] = 0;  // zero pad
}

// ---------------------------------------------------------------------------
// GEMM + fused bias/BN/ReLU via v_wmma_f32_16x16x32_bf16.
// X:  [M, lda] bf16, lda = RUP32(Cin), rows zero-padded; M % 64 == 0.
// Wp: [ldc, ldk] bf16, pre-transposed + zero-padded -> guard-free staging.
// Block = 4 waves; B tile block-shared; each wave computes 64x16 D
// (4 subtiles, 4 WMMAs per K-step). Y: [M, ldc], pad columns zeroed.
// ---------------------------------------------------------------------------
__global__ void __launch_bounds__(GEMM_WAVES * 32)
gemm_bn_relu_kernel(const unsigned short* __restrict__ X,
                    const unsigned short* __restrict__ Wp,
                    const float* __restrict__ bias,
                    const float* __restrict__ gamma,
                    const float* __restrict__ beta,
                    const float* __restrict__ mean,
                    const float* __restrict__ var,
                    unsigned short* __restrict__ Y,
                    int M, int Cin, int Cout, int lda, int ldk, int ldc) {
  __shared__ __align__(16) unsigned Bs[256];              // [n(16)][kd(16)]
  __shared__ __align__(16) unsigned As[GEMM_WAVES][NSUB * 256];
  const int wave = threadIdx.x >> 5;
  const int lane = threadIdx.x & 31;
  const int half = lane >> 4;
  const int l15  = lane & 15;
  int mbase = (blockIdx.x * GEMM_WAVES + wave) * (16 * NSUB);
  const bool active = mbase < M;
  if (!active) mbase = 0;                                 // stay in-bounds, skip store
  const int n0 = blockIdx.y * 16;
  v8f acc[NSUB] = {};
  for (int k0 = 0; k0 < Cin; k0 += 32) {
    __syncthreads();
    // ---- stage B: one aligned uint2 per thread from packed bf16 weights ----
    {
      int c = threadIdx.x;               // 128 uint2 chunks = 256 dwords
      int nn = c >> 3, kd2 = c & 7;
      const uint2* wr = (const uint2*)(Wp + (size_t)(n0 + nn) * ldk + k0);
      *(uint2*)&Bs[nn * 16 + kd2 * 2] = wr[kd2];
    }
    // ---- stage A: guard-free 16B vector loads (rows zero-padded) ----
#pragma unroll
    for (int j = 0; j < NSUB * 2; ++j) {
      int cidx = lane + 32 * j;          // uint4 chunk id, 0 .. NSUB*64-1
      int sub = cidx >> 6, tt = cidx & 63;
      int row = tt >> 2, kq = tt & 3;
      int gm = mbase + sub * 16 + row;
      const uint4* xr = (const uint4*)(X + (size_t)gm * lda + k0);
      *(uint4*)&As[wave][sub * 256 + row * 16 + kq * 4] = xr[kq];
    }
    if (k0 + 32 < Cin) {  // speculative prefetch of next K tile -> global_prefetch_b8
      __builtin_prefetch(X + (size_t)(mbase + l15) * lda + k0 + 32, 0, 1);
      __builtin_prefetch(Wp + (size_t)(n0 + l15) * ldk + k0 + 32, 0, 1);
    }
    __syncthreads();
    // ---- fragments: CDNA5 16-bit swizzle, dword d = (i<4?0:8)+half*4+(i&3) ----
    Frag fb;
#pragma unroll
    for (int i = 0; i < 8; ++i) {
      int d = ((i < 4) ? 0 : 8) + half * 4 + (i & 3);
      fb.u[i] = Bs[l15 * 16 + d];
    }
#pragma unroll
    for (int sub = 0; sub < NSUB; ++sub) {
      Frag fa;
#pragma unroll
      for (int i = 0; i < 8; ++i) {
        int d = ((i < 4) ? 0 : 8) + half * 4 + (i & 3);
        fa.u[i] = As[wave][sub * 256 + l15 * 16 + d];
      }
      acc[sub] = __builtin_amdgcn_wmma_f32_16x16x32_bf16(false, fa.v, false, fb.v,
                                                         (short)0, acc[sub], false, false);
    }
  }
  // ---- epilogue: y = (x@W + b - mean)*gamma*rsqrt(var+eps) + beta, ReLU ----
  int n = n0 + l15;
  int cn = imin(n, Cout - 1);
  float s = gamma[cn] * rsqrtf(var[cn] + 1e-5f);
  float c = (bias[cn] - mean[cn]) * s + beta[cn];
  bool valid = n < Cout;
  if (active) {
#pragma unroll
    for (int sub = 0; sub < NSUB; ++sub) {
#pragma unroll
      for (int r = 0; r < 8; ++r) {
        int m = mbase + sub * 16 + r + half * 8;  // D: VGPR r -> M=r+8*half, N=l15
        float v = valid ? fmaxf(fmaf(acc[sub][r], s, c), 0.f) : 0.f;
        Y[(size_t)m * ldc + n] = f2bf_u(v);
      }
    }
  }
}

// ---------------------------------------------------------------------------
// Max-pool over K samples, write into concat buffer at channel offset
// (input row stride == C: every scale-final cout is a multiple of 32)
// ---------------------------------------------------------------------------
__global__ void maxpool_kernel(const unsigned short* __restrict__ Yin,
                               unsigned short* __restrict__ out,
                               int K, int C, int Ctot, int coff, int total) {
  int t = blockIdx.x * blockDim.x + threadIdx.x;
  if (t >= total) return;
  int c = t % C, row = t / C;
  const unsigned short* p = Yin + (size_t)row * K * C + c;
  float m = -1e30f;
  for (int k = 0; k < K; ++k) m = fmaxf(m, bfu2f(p[(size_t)k * C]));
  out[(size_t)row * Ctot + coff + c] = f2bf_u(m);
}

// ---------------------------------------------------------------------------
// Feature propagation: 3-NN inverse-distance interpolation + concat pts1
// (C1+C2 is a multiple of 32 for every FP stage -> no pad needed)
// ---------------------------------------------------------------------------
__global__ void interp_kernel(const float* __restrict__ xyz1, const float* __restrict__ xyz2,
                              const unsigned short* __restrict__ pts1,
                              const unsigned short* __restrict__ pts2,
                              unsigned short* __restrict__ X,
                              int N1, int S, int C1, int C2, int total) {
  int t = blockIdx.x * blockDim.x + threadIdx.x;
  if (t >= total) return;
  int b = t / N1;
  const float* p = xyz1 + (size_t)t * 3;
  const float* q = xyz2 + (size_t)b * S * 3;
  float d0 = 1e30f, d1 = 1e30f, d2 = 1e30f;
  int i0 = 0, i1 = 0, i2 = 0;
  for (int s = 0; s < S; ++s) {
    float dx = p[0] - q[s * 3], dy = p[1] - q[s * 3 + 1], dz = p[2] - q[s * 3 + 2];
    float d = dx * dx + dy * dy + dz * dz;
    if (d < d0)      { d2 = d1; i2 = i1; d1 = d0; i1 = i0; d0 = d; i0 = s; }
    else if (d < d1) { d2 = d1; i2 = i1; d1 = d; i1 = s; }
    else if (d < d2) { d2 = d; i2 = s; }
  }
  float w0 = 1.f / (d0 + 1e-8f), w1 = 1.f / (d1 + 1e-8f), w2 = 1.f / (d2 + 1e-8f);
  float ws = w0 + w1 + w2; w0 /= ws; w1 /= ws; w2 /= ws;
  unsigned short* orow = X + (size_t)t * (C1 + C2);
  if (pts1) {
    const unsigned short* r = pts1 + (size_t)t * C1;
    for (int c = 0; c < C1; ++c) orow[c] = r[c];
  }
  const unsigned short* base = pts2 + (size_t)b * S * C2;
  const unsigned short* r0 = base + (size_t)i0 * C2;
  const unsigned short* r1 = base + (size_t)i1 * C2;
  const unsigned short* r2 = base + (size_t)i2 * C2;
  for (int c = 0; c < C2; ++c) {
    float v = w0 * bfu2f(r0[c]) + w1 * bfu2f(r1[c]) + w2 * bfu2f(r2[c]);
    orow[C1 + c] = f2bf_u(v);
  }
}

// ---------------------------------------------------------------------------
// Head: path/kpt 128->2 projections, output [B,2,N] fp32 (path then kpt)
// ---------------------------------------------------------------------------
__global__ void head_out_kernel(const unsigned short* __restrict__ x,
                                const float* __restrict__ Wp, const float* __restrict__ bp,
                                const float* __restrict__ Wk, const float* __restrict__ bk,
                                float* __restrict__ out, int N, int total) {
  int t = blockIdx.x * blockDim.x + threadIdx.x;
  if (t >= total) return;
  int n = t % N, b = t / N;
  const unsigned short* r = x + (size_t)t * 128;
  float p0 = bp[0], p1 = bp[1], k0 = bk[0], k1 = bk[1];
  for (int c = 0; c < 128; ++c) {
    float v = bfu2f(r[c]);
    p0 = fmaf(v, Wp[c * 2], p0);     p1 = fmaf(v, Wp[c * 2 + 1], p1);
    k0 = fmaf(v, Wk[c * 2], k0);     k1 = fmaf(v, Wk[c * 2 + 1], k1);
  }
  out[(size_t)(b * 2 + 0) * N + n] = p0;
  out[(size_t)(b * 2 + 1) * N + n] = p1;
  float* ok = out + (size_t)total * 2;
  ok[(size_t)(b * 2 + 0) * N + n] = k0;
  ok[(size_t)(b * 2 + 1) * N + n] = k1;
}

// ---------------------------------------------------------------------------
// Host orchestration
// ---------------------------------------------------------------------------
struct LayerP {
  const float *W, *b, *g, *be, *mu, *va;
  int cin, cout;
  unsigned short* Wp;   // packed bf16 weights [ldc][ldk]
  int ldk, ldc;
};

static void run_gemm(const unsigned short* X, const LayerP& L, unsigned short* Y,
                     int M, hipStream_t s) {
  int lda = RUP32(L.cin);
  dim3 grid(CEILDIV(CEILDIV(M, 16 * NSUB), GEMM_WAVES), L.ldc / 16);
  gemm_bn_relu_kernel<<<grid, GEMM_WAVES * 32, 0, s>>>(
      X, L.Wp, L.b, L.g, L.be, L.mu, L.va, Y, M, L.cin, L.cout, lda, L.ldk, L.ldc);
}

extern "C" void kernel_launch(void* const* d_in, const int* in_sizes, int n_in,
                              void* d_out, int out_size, void* d_ws, size_t ws_size,
                              hipStream_t stream) {
  (void)in_sizes; (void)n_in; (void)out_size; (void)ws_size;
  const int Bc = 2, N0 = 16384;

  // ---- walk params in setup_inputs() insertion order ----
  int pi = 0;
  const float* xyz_in = (const float*)d_in[pi++];
  auto nf = [&]() { return (const float*)d_in[pi++]; };
  auto load_layer = [&](int cin, int cout) {
    LayerP L; L.W = nf(); L.b = nf(); L.g = nf(); L.be = nf(); L.mu = nf(); L.va = nf();
    L.cin = cin; L.cout = cout; L.Wp = nullptr; L.ldk = RUP32(cin); L.ldc = RUP32(cout);
    return L;
  };

  static const int   sa_np[4]        = {1024, 256, 64, 16};
  static const float sa_r[4][2]      = {{0.05f, 0.1f}, {0.1f, 0.2f}, {0.2f, 0.4f}, {0.4f, 0.8f}};
  static const int   sa_K[2]         = {16, 32};
  static const int   sa_inch[4]      = {6, 96, 256, 512};
  static const int   sa_mlps[4][2][3] = {
      {{16, 16, 32},   {32, 32, 64}},
      {{64, 64, 128},  {64, 96, 128}},
      {{128, 196, 256},{128, 196, 256}},
      {{256, 256, 512},{256, 384, 512}}};
  static const int fp_in[4]    = {1536, 512, 352, 128};   // fp4, fp3, fp2, fp1
  static const int fp_n[4]     = {2, 2, 2, 3};
  static const int fp_mlp[4][3] = {{256, 256, 0}, {256, 256, 0}, {256, 128, 0}, {128, 128, 128}};

  LayerP saL[4][2][3];
  for (int i = 0; i < 4; ++i)
    for (int s = 0; s < 2; ++s) {
      int cin = sa_inch[i] + 3;
      for (int j = 0; j < 3; ++j) { saL[i][s][j] = load_layer(cin, sa_mlps[i][s][j]); cin = sa_mlps[i][s][j]; }
    }
  LayerP fpL[4][3];
  for (int i = 0; i < 4; ++i) {
    int cin = fp_in[i];
    for (int j = 0; j < fp_n[i]; ++j) { fpL[i][j] = load_layer(cin, fp_mlp[i][j]); cin = fp_mlp[i][j]; }
  }
  LayerP hL = [&]() {
    LayerP L; L.W = nf(); L.b = nf(); L.g = nf(); L.be = nf(); L.mu = nf(); L.va = nf();
    L.cin = 128; L.cout = 128; L.Wp = nullptr; L.ldk = 128; L.ldc = 128; return L;
  }();
  const float* pW = nf(); const float* pb = nf();
  const float* kW = nf(); const float* kb = nf();

  // ---- workspace bump allocator ----
  char* ws = (char*)d_ws; size_t off = 0;
  auto alloc = [&](size_t bytes) { off = (off + 255) & ~(size_t)255; void* p = ws + off; off += bytes; return p; };

  int nper[5]  = {N0, 1024, 256, 64, 16};
  int cfeat[5] = {6, 96, 256, 512, 1024};
  float* xyzL[5]; unsigned short* featL[5];
  for (int l = 0; l < 5; ++l) {
    xyzL[l]  = (float*)alloc((size_t)Bc * nper[l] * 3 * sizeof(float));
    featL[l] = (unsigned short*)alloc((size_t)Bc * nper[l] * cfeat[l] * sizeof(unsigned short));
  }
  int* fpsIdx = (int*)alloc((size_t)Bc * 1024 * sizeof(int));
  int* gidx   = (int*)alloc((size_t)Bc * 1024 * 32 * sizeof(int));
  unsigned short* gA = (unsigned short*)alloc((size_t)16 << 20);
  unsigned short* gB = (unsigned short*)alloc((size_t)16 << 20);
  unsigned short* gC = (unsigned short*)alloc((size_t)16 << 20);
  unsigned short* l3n = (unsigned short*)alloc((size_t)Bc * 64 * 256 * 2);
  unsigned short* l2n = (unsigned short*)alloc((size_t)Bc * 256 * 256 * 2);
  unsigned short* l1n = (unsigned short*)alloc((size_t)Bc * 1024 * 128 * 2);
  unsigned short* l0n = (unsigned short*)alloc((size_t)Bc * N0 * 128 * 2);

  // ---- pack all weights (once per call; tiny vs. GEMM reuse) ----
  auto pack_layer = [&](LayerP& L) {
    int total = L.ldc * L.ldk;
    L.Wp = (unsigned short*)alloc((size_t)total * sizeof(unsigned short));
    pack_w_kernel<<<CEILDIV(total, 256), 256, 0, stream>>>(L.W, L.Wp, L.cin, L.cout, L.ldk, total);
  };
  for (int i = 0; i < 4; ++i)
    for (int s = 0; s < 2; ++s)
      for (int j = 0; j < 3; ++j) pack_layer(saL[i][s][j]);
  for (int i = 0; i < 4; ++i)
    for (int j = 0; j < fp_n[i]; ++j) pack_layer(fpL[i][j]);
  pack_layer(hL);

  // ---- l0 prep ----
  prep_kernel<<<CEILDIV(Bc * N0, 256), 256, 0, stream>>>(xyz_in, xyzL[0], featL[0], N0, Bc * N0);

  // ---- SA stages ----
  for (int lvl = 0; lvl < 4; ++lvl) {
    int Nin = nper[lvl], np = sa_np[lvl], C0 = cfeat[lvl];
    size_t shmem = (size_t)Nin * sizeof(float) + 256 * (sizeof(float) + sizeof(int));
    fps_kernel<<<Bc, 256, shmem, stream>>>(xyzL[lvl], Nin, np, fpsIdx);
    gather_xyz_kernel<<<CEILDIV(Bc * np, 256), 256, 0, stream>>>(xyzL[lvl], fpsIdx, xyzL[lvl + 1],
                                                                 Nin, np, Bc * np);
    int coff = 0;
    for (int s = 0; s < 2; ++s) {
      int K = sa_K[s];
      float r = sa_r[lvl][s];
      ball_query_kernel<<<CEILDIV(Bc * np, 128), 128, 0, stream>>>(
          xyzL[lvl], xyzL[lvl + 1], gidx, Nin, np, K, r * r, Bc * np);
      int Mrows = Bc * np * K;
      int lda0 = RUP32(C0 + 3);
      group_kernel<<<CEILDIV(Mrows, 256), 256, 0, stream>>>(
          featL[lvl], xyzL[lvl], xyzL[lvl + 1], gidx, gA, Nin, np, K, C0, lda0, Mrows);
      const unsigned short* cur = gA;
      for (int j = 0; j < 3; ++j) {
        unsigned short* out = (j == 2) ? gC : ((cur == gA) ? gB : gA);
        run_gemm(cur, saL[lvl][s][j], out, Mrows, stream);
        cur = out;
      }
      int cout = sa_mlps[lvl][s][2];   // multiple of 32 -> ldc == cout
      maxpool_kernel<<<CEILDIV(Bc * np * cout, 256), 256, 0, stream>>>(
          gC, featL[lvl + 1], K, cout, cfeat[lvl + 1], coff, Bc * np * cout);
      coff += cout;
    }
  }

  // ---- FP stages ----
  auto run_fp = [&](int i, const float* x1, const float* x2,
                    const unsigned short* p1, int C1, const unsigned short* p2, int C2,
                    int N1, int S, unsigned short* persist) {
    int total = Bc * N1;
    interp_kernel<<<CEILDIV(total, 128), 128, 0, stream>>>(x1, x2, p1, p2, gA, N1, S, C1, C2, total);
    const unsigned short* cur = gA;
    for (int j = 0; j < fp_n[i]; ++j) {
      unsigned short* out = (j == fp_n[i] - 1) ? persist : ((cur == gA) ? gB : gA);
      run_gemm(cur, fpL[i][j], out, total, stream);
      cur = out;
    }
  };
  run_fp(0, xyzL[3], xyzL[4], featL[3], 512, featL[4], 1024, 64, 16, l3n);
  run_fp(1, xyzL[2], xyzL[3], featL[2], 256, l3n, 256, 256, 64, l2n);
  run_fp(2, xyzL[1], xyzL[2], featL[1], 96,  l2n, 256, 1024, 256, l1n);
  run_fp(3, xyzL[0], xyzL[1], (const unsigned short*)nullptr, 0, l1n, 128, N0, 1024, l0n);

  // ---- head ----
  run_gemm(l0n, hL, gA, Bc * N0, stream);
  head_out_kernel<<<CEILDIV(Bc * N0, 128), 128, 0, stream>>>(
      gA, pW, pb, kW, kb, (float*)d_out, N0, Bc * N0);
}
